// Attention_87548613362175
// MI455X (gfx1250) — matrix operand order repaired and checked by reading, not compiled, not run on hardware
//
#include <hip/hip_runtime.h>
#include <hip/hip_bf16.h>

// Problem constants (match reference)
#define B_  32
#define T_  4096
#define D_  1024
#define H_  8
#define HD_ 128
#define P_  1024
#define TT_ 128                 // t-tile rows per workgroup
#define NT_ (T_ / TT_)          // 32 tiles along T
#define XS_ 1032                // LDS row stride (bf16 elems), padded to spread banks

typedef __attribute__((ext_vector_type(16))) __bf16 v16bf;
typedef __attribute__((ext_vector_type(8)))  __bf16 v8bf;
typedef __attribute__((ext_vector_type(8)))  float  v8f;

static __device__ __forceinline__ unsigned short f2bf(float f) {
  unsigned int u = __float_as_uint(f);
  u = u + 0x7FFFu + ((u >> 16) & 1u);      // round-to-nearest-even
  return (unsigned short)(u >> 16);
}
static __device__ __forceinline__ float bf2f(unsigned short h) {
  return __uint_as_float(((unsigned int)h) << 16);
}

// ---------------- prep: psi^T in bf16 (L2-resident, 2MB) ----------------
__global__ void prep_psiT(const float* __restrict__ psi_w,
                          unsigned short* __restrict__ psiT) {
  int idx = blockIdx.x * blockDim.x + threadIdx.x;   // idx = d*P + p
  if (idx >= P_ * D_) return;
  int d = idx / P_, p = idx % P_;
  psiT[idx] = f2bf(psi_w[p * D_ + d]);
}

// ---------------- prep: q = relu(dec @ phi^T + b), exact fp32 ----------------
__global__ void prep_q(const float* __restrict__ dec, const float* __restrict__ phi_w,
                       const float* __restrict__ phi_b, float* __restrict__ q) {
  int idx = blockIdx.x * blockDim.x + threadIdx.x;   // B*P
  if (idx >= B_ * P_) return;
  int b = idx / P_, p = idx % P_;
  const float4* x4 = (const float4*)(dec + b * D_);
  const float4* w4 = (const float4*)(phi_w + p * D_);
  float s = 0.f;
  for (int i = 0; i < D_ / 4; ++i) {
    float4 xv = x4[i], wv = w4[i];
    s += xv.x * wv.x + xv.y * wv.y + xv.z * wv.z + xv.w * wv.w;
  }
  s += phi_b[p];
  q[idx] = fmaxf(s, 0.f);
}

// ---------------- fused MLP-GEMM + energy + tile softmax + context partials ----------------
__global__ void __launch_bounds__(256, 1)
attn_main(const float* __restrict__ x,              // [B,T,D] fp32
          const unsigned short* __restrict__ psiT,  // [D,P]  bf16
          const float* __restrict__ psi_b,          // [P]
          const float* __restrict__ q,              // [B,P]
          float* __restrict__ e7,                   // [B,T] raw energy, head 7
          float* __restrict__ part)                 // [B,NT,H,130]: m, l, c[128]
{
  extern __shared__ unsigned char smem[];
  unsigned short* xbf = (unsigned short*)smem;               // [TT_][XS_] bf16
  float* eng  = (float*)(smem + TT_ * XS_ * 2);              // [2][TT_]
  float* pv   = eng + 2 * TT_;                               // [TT_]
  float* red  = pv + TT_;                                    // [256]
  float* cbuf = red + 256;                                   // [256]

  const int tid  = threadIdx.x;
  const int b    = blockIdx.y;
  const int tile = blockIdx.x;
  const int t0   = tile * TT_;

  // Phase 1: stream x tile (fp32) -> bf16 in LDS.  Single HBM pass over x.
  const float* xg = x + ((size_t)b * T_ + t0) * D_;
  for (int i = tid; i < TT_ * (D_ / 4); i += 256) {
    int r  = i >> 8;            // D_/4 == 256 float4 per row
    int c4 = i & 255;
    float4 v = ((const float4*)(xg + (size_t)r * D_))[c4];
    unsigned short* dst = xbf + r * XS_ + c4 * 4;
    dst[0] = f2bf(v.x); dst[1] = f2bf(v.y); dst[2] = f2bf(v.z); dst[3] = f2bf(v.w);
  }
  __syncthreads();

  const int lane = tid & 31;
  const int wv   = tid >> 5;      // wave 0..7
  const int pl   = lane & 15;
  const int half = lane >> 4;
  const int wm   = wv & 3;        // M group: rows wm*32 .. wm*32+31
  const int wn   = wv >> 2;       // N group: cols wn*64 .. wn*64+63

  const unsigned short* aBase = xbf + (wm * 32 + pl) * XS_ + half * 8;

  for (int h = 0; h < H_; ++h) {
    v8f acc[2][4] = {};           // 2 M-tiles x 4 N-tiles of 16x16 f32

    // B fragment base: B[k][n] = psiT[(k0+k)*P + h*HD + wn*64 + n], lane n = pl, k = 8*half (+16)
    const unsigned short* bB = psiT + (size_t)(half * 8) * P_ + h * HD_ + wn * 64 + pl;

    // fragment loaders -----------------------------------------------------
    auto loadB4 = [&](int kk, v16bf Bv[4]) {
      const unsigned short* bp = bB + (size_t)kk * P_;
      #pragma unroll
      for (int c = 0; c < 4; ++c)
        for (int s = 0; s < 16; ++s)
          Bv[c][s] = __builtin_bit_cast(__bf16, bp[c * 16 + (size_t)(s < 8 ? s : s + 8) * P_]);
    };
    auto loadA2 = [&](int kk, v16bf Af[2]) {
      // 16-bit A 16x32 layout: lanes 0-15 K 0-7/16-23, lanes 16-31 K 8-15/24-31
      #pragma unroll
      for (int m2 = 0; m2 < 2; ++m2) {
        const unsigned short* ap = aBase + m2 * (16 * XS_) + kk;
        v8bf lo = *(const v8bf*)ap;
        v8bf hi = *(const v8bf*)(ap + 16);
        Af[m2] = __builtin_shufflevector(lo, hi,
                   0,1,2,3,4,5,6,7,8,9,10,11,12,13,14,15);
      }
    };

    // double-buffered k-loop: a full stage of loads is in flight ahead of
    // the dependent WMMA chain (one WG/WGP -> little wave-level hiding).
    v16bf B0[4], B1[4], A0[2], A1[2];
    loadB4(0, B0);
    loadA2(0, A0);
    for (int k0 = 0; k0 < D_; k0 += 64) {
      loadB4(k0 + 32, B1);
      loadA2(k0 + 32, A1);
      if (k0 + 64 < D_)
        __builtin_prefetch(bB + (size_t)(k0 + 64) * P_, 0, 1);
      #pragma unroll
      for (int c = 0; c < 4; ++c)
        #pragma unroll
        for (int m2 = 0; m2 < 2; ++m2)
          acc[m2][c] = __builtin_amdgcn_wmma_f32_16x16x32_bf16(
              false, A0[m2], false, B0[c], (short)0, acc[m2][c], false, false);
      if (k0 + 64 < D_) {
        loadB4(k0 + 64, B0);
        loadA2(k0 + 64, A0);
      }
      #pragma unroll
      for (int c = 0; c < 4; ++c)
        #pragma unroll
        for (int m2 = 0; m2 < 2; ++m2)
          acc[m2][c] = __builtin_amdgcn_wmma_f32_16x16x32_bf16(
              false, A1[m2], false, B1[c], (short)0, acc[m2][c], false, false);
    }

    // bias + relu + q-weighted reduction over this head's 128 features
    float qv[4], bv[4];
    #pragma unroll
    for (int c = 0; c < 4; ++c) {
      int col = h * HD_ + wn * 64 + c * 16 + pl;
      qv[c] = q[b * P_ + col];
      bv[c] = psi_b[col];
    }
    float pj[2][8];
    #pragma unroll
    for (int m2 = 0; m2 < 2; ++m2)
      #pragma unroll
      for (int j = 0; j < 8; ++j) {
        float s = 0.f;
        #pragma unroll
        for (int c = 0; c < 4; ++c)
          s += fmaxf(acc[m2][c][j] + bv[c], 0.f) * qv[c];
        // C layout: VGPR j, lanes 0-15 -> row j, lanes 16-31 -> row j+8; N = pl
        s += __shfl_xor(s, 1, 32);
        s += __shfl_xor(s, 2, 32);
        s += __shfl_xor(s, 4, 32);
        s += __shfl_xor(s, 8, 32);   // masks stay within 16-lane half-groups
        pj[m2][j] = s;
      }
    if (pl == 0) {
      #pragma unroll
      for (int m2 = 0; m2 < 2; ++m2)
        #pragma unroll
        for (int j = 0; j < 8; ++j)
          eng[wn * TT_ + wm * 32 + m2 * 16 + half * 8 + j] = pj[m2][j];
    }
    __syncthreads();

    // tile-local softmax stats (flash-decoding partials)
    float e = 0.f;
    if (tid < TT_) e = eng[tid] + eng[TT_ + tid];   // combine the two N-halves
    red[tid] = (tid < TT_) ? e : -3.4e38f;
    __syncthreads();
    for (int s = 128; s > 0; s >>= 1) {
      if (tid < s) red[tid] = fmaxf(red[tid], red[tid + s]);
      __syncthreads();
    }
    float mloc = red[0];
    __syncthreads();
    float p = 0.f;
    if (tid < TT_) { p = __expf(e - mloc); pv[tid] = p; }
    red[tid] = (tid < TT_) ? p : 0.f;
    __syncthreads();
    for (int s = 128; s > 0; s >>= 1) {
      if (tid < s) red[tid] += red[tid + s];
      __syncthreads();
    }
    float lsum = red[0];
    if (h == H_ - 1 && tid < TT_)
      e7[(size_t)b * T_ + t0 + tid] = e;            // raw energy, normalized later

    // partial context: c[d] = sum_t p[t] * x_bf16[t][h*HD+d]
    {
      int d  = tid & 127;
      int tg = tid >> 7;
      float s = 0.f;
      const unsigned short* col = xbf + h * HD_ + d;
      for (int t = tg * 64; t < tg * 64 + 64; ++t)
        s += pv[t] * bf2f(col[t * XS_]);
      cbuf[tid] = s;
      __syncthreads();
      float* rec = part + ((((size_t)b * NT_ + tile) * H_) + h) * 130;
      if (tid < 128) rec[2 + tid] = cbuf[tid] + cbuf[tid + 128];
      if (tid == 0) { rec[0] = mloc; rec[1] = lsum; }
    }
    __syncthreads();
  }
}

// ---------------- combine tile partials per (b,h) ----------------
__global__ void attn_reduce(const float* __restrict__ part,
                            float* __restrict__ ctx_out,   // [B,D]
                            float* __restrict__ stats) {   // [B,2] for head 7
  int bh = blockIdx.x;
  int b = bh / H_, h = bh % H_;
  int tid = threadIdx.x;                   // 128
  __shared__ float mt[NT_];
  __shared__ float red[128];
  const size_t tstride = (size_t)H_ * 130;
  const float* base = part + (((size_t)b * NT_) * H_ + h) * 130;
  if (tid < NT_) mt[tid] = base[tid * tstride];
  __syncthreads();
  float m = -3.4e38f;
  for (int i = 0; i < NT_; ++i) m = fmaxf(m, mt[i]);
  float ls = 0.f;
  if (tid < NT_) ls = base[tid * tstride + 1] * __expf(mt[tid] - m);
  red[tid] = ls;
  __syncthreads();
  for (int s = 64; s > 0; s >>= 1) {
    if (tid < s) red[tid] += red[tid + s];
    __syncthreads();
  }
  float lsum = red[0];
  float c = 0.f;
  for (int i = 0; i < NT_; ++i)
    c += base[i * tstride + 2 + tid] * __expf(mt[i] - m);
  ctx_out[(size_t)b * D_ + h * HD_ + tid] = c / lsum;
  if (h == H_ - 1 && tid == 0) { stats[b * 2] = m; stats[b * 2 + 1] = lsum; }
}

// ---------------- normalize head-7 scores ----------------
__global__ void attn_score(const float* __restrict__ e7, const float* __restrict__ stats,
                           float* __restrict__ score_out) {
  int i = blockIdx.x * blockDim.x + threadIdx.x;
  if (i >= B_ * T_) return;
  int b = i / T_;
  score_out[i] = __expf(e7[i] - stats[b * 2]) / stats[b * 2 + 1];
}

extern "C" void kernel_launch(void* const* d_in, const int* in_sizes, int n_in,
                              void* d_out, int out_size, void* d_ws, size_t ws_size,
                              hipStream_t stream) {
  const float* dec  = (const float*)d_in[0];   // [B,1,D]
  const float* lis  = (const float*)d_in[1];   // [B,T,D]
  const float* phiw = (const float*)d_in[2];   // [P,D]
  const float* phib = (const float*)d_in[3];   // [P]
  const float* psiw = (const float*)d_in[4];   // [P,D]
  const float* psib = (const float*)d_in[5];   // [P]

  float* out       = (float*)d_out;
  float* score_out = out;                        // [B,T]  (head H-1 scores)
  float* ctx_out   = out + (size_t)B_ * T_;      // [B,D]

  unsigned char* ws = (unsigned char*)d_ws;
  size_t off = 0;
  float* qbuf = (float*)(ws + off);            off += (size_t)B_ * P_ * 4;      // 128 KB
  unsigned short* psiT = (unsigned short*)(ws + off); off += (size_t)P_ * D_ * 2; // 2 MB
  float* e7 = (float*)(ws + off);              off += (size_t)B_ * T_ * 4;      // 512 KB
  float* part = (float*)(ws + off);            off += (size_t)B_ * NT_ * H_ * 130 * 4;
  float* stats = (float*)(ws + off);           off += (size_t)B_ * 2 * 4;

  prep_psiT<<<(P_ * D_ + 255) / 256, 256, 0, stream>>>(psiw, psiT);
  prep_q<<<(B_ * P_ + 255) / 256, 256, 0, stream>>>(dec, phiw, phib, qbuf);

  dim3 grid(NT_, B_);
  size_t smem = (size_t)TT_ * XS_ * 2 + (size_t)(2 * TT_ + TT_ + 256 + 256) * 4;
  attn_main<<<grid, 256, smem, stream>>>(lis, psiT, psib, qbuf, e7, part);

  attn_reduce<<<B_ * H_, 128, 0, stream>>>(part, ctx_out, stats);
  attn_score<<<(B_ * T_ + 255) / 256, 256, 0, stream>>>(e7, stats, score_out);
}
